// BailingMoeBlock_80522046865498
// MI455X (gfx1250) — matrix operand order repaired and checked by reading, not compiled, not run on hardware
//
#include <hip/hip_runtime.h>

typedef __attribute__((ext_vector_type(16))) __bf16 v16bf;
typedef __attribute__((ext_vector_type(8)))  float  v8f;
typedef __attribute__((ext_vector_type(4)))  int    v4i;
typedef __attribute__((address_space(1))) v4i gv4i;   // global int4
typedef __attribute__((address_space(3))) v4i lv4i;   // LDS int4

#define T_TOK 1024
#define HDIM  2048
#define NH    16
#define NKV   4
#define HD    128
#define QKVN  3072
#define NEXP  16
#define EI    512
#define SI    1024

#if defined(__has_builtin)
#if __has_builtin(__builtin_amdgcn_global_load_async_to_lds_b128)
#define ASYNC_COPY 1
#endif
#endif

static __device__ __forceinline__ unsigned short f2bf(float f) {
  unsigned int u = __float_as_uint(f);
  u += 0x7FFFu + ((u >> 16) & 1u);
  return (unsigned short)(u >> 16);
}
static __device__ __forceinline__ float bf2f(unsigned short h) {
  return __uint_as_float(((unsigned int)h) << 16);
}
static __device__ __forceinline__ v8f vzero() {
  v8f z;
#pragma unroll
  for (int i = 0; i < 8; ++i) z[i] = 0.0f;
  return z;
}
// Fill a 16xbf16 fragment from two contiguous 8-element (16B) runs.
static __device__ __forceinline__ v16bf ldfrag(const unsigned short* p0, const unsigned short* p1) {
  v16bf f;
  ((uint4*)&f)[0] = *(const uint4*)p0;
  ((uint4*)&f)[1] = *(const uint4*)p1;
  return f;
}
static __device__ __forceinline__ v8f wmma_bf16(v16bf a, v16bf b, v8f c) {
  return __builtin_amdgcn_wmma_f32_16x16x32_bf16(false, a, false, b, (short)0, c, false, false);
}

// 16B global -> LDS copy. Async (ASYNCcnt) on CDNA5, sync fallback otherwise.
static __device__ __forceinline__ void cp16(unsigned short* lds, const unsigned short* g) {
#ifdef ASYNC_COPY
  __builtin_amdgcn_global_load_async_to_lds_b128(
      (gv4i*)(unsigned long long)g,
      (lv4i*)(unsigned int)(unsigned long long)lds,
      0, 0);
#else
  *(uint4*)lds = *(const uint4*)g;
#endif
}
static __device__ __forceinline__ void wait_cp() {
#ifdef ASYNC_COPY
#if __has_builtin(__builtin_amdgcn_s_wait_asynccnt)
  __builtin_amdgcn_s_wait_asynccnt(0);
#else
  asm volatile("s_wait_asynccnt 0x0" ::: "memory");
#endif
#endif
}

// ---------------- small utility kernels ----------------

__global__ void k_zero_i32(int* p, int n) {
  int i = blockIdx.x * blockDim.x + threadIdx.x;
  if (i < n) p[i] = 0;
}

// dst[N][K] (bf16, K-major) = transpose of src[K][N] (f32); blockIdx.z = matrix id
__global__ __launch_bounds__(256) void k_cvt_t(const float* __restrict__ src,
                                               unsigned short* __restrict__ dst,
                                               int K, int N) {
  __shared__ float tile[32][33];
  const size_t mat = (size_t)blockIdx.z * (size_t)K * (size_t)N;
  const int k0 = blockIdx.y * 32, n0 = blockIdx.x * 32;
  const int tx = threadIdx.x & 31, ty = threadIdx.x >> 5;
#pragma unroll
  for (int i = 0; i < 4; ++i)
    tile[ty + i * 8][tx] = src[mat + (size_t)(k0 + ty + i * 8) * N + n0 + tx];
  __syncthreads();
#pragma unroll
  for (int i = 0; i < 4; ++i)
    dst[mat + (size_t)(n0 + ty + i * 8) * K + k0 + tx] = f2bf(tile[tx][ty + i * 8]);
}

// sum = a + b (kept fp32), xb = bf16(rmsnorm(sum) * w)
__global__ __launch_bounds__(256) void k_add_rms(const float* __restrict__ a,
                                                 const float* __restrict__ b,
                                                 const float* __restrict__ w,
                                                 float* __restrict__ sum_out,
                                                 unsigned short* __restrict__ xb) {
  __shared__ float red[256];
  const int t = blockIdx.x, tid = threadIdx.x;
  float loc = 0.f, vals[8];
#pragma unroll
  for (int i = 0; i < 8; ++i) {
    int c = tid + i * 256;
    float v = a[(size_t)t * HDIM + c] + b[(size_t)t * HDIM + c];
    vals[i] = v;
    sum_out[(size_t)t * HDIM + c] = v;
    loc += v * v;
  }
  red[tid] = loc;
  __syncthreads();
  for (int s = 128; s > 0; s >>= 1) { if (tid < s) red[tid] += red[tid + s]; __syncthreads(); }
  float rs = rsqrtf(red[0] / (float)HDIM + 1e-6f);
#pragma unroll
  for (int i = 0; i < 8; ++i) {
    int c = tid + i * 256;
    xb[(size_t)t * HDIM + c] = f2bf(vals[i] * rs * w[c]);
  }
}

// ---------------- generic bf16 WMMA GEMM: C(f32) = A x Bt^T ----------------
// A: MxK row-major bf16. Bt: NxK row-major bf16 (pre-transposed weights).
// Tile 128x128x32, 8 waves, double-buffered async LDS staging.
__global__ __launch_bounds__(256) void k_gemm(const unsigned short* __restrict__ Ag,
                                              const unsigned short* __restrict__ Bt,
                                              float* __restrict__ C, int N, int K) {
  __shared__ unsigned short As[2][128 * 32];
  __shared__ unsigned short Bs[2][128 * 32];   // [n][k], K contiguous
  const int tid = threadIdx.x;
  const int lane = tid & 31, wave = tid >> 5;
  const int ln = lane & 15, half = lane >> 4;
  const int wm = wave >> 1, wn = wave & 1;
  const int m0 = blockIdx.y * 128, n0 = blockIdx.x * 128;
  const int r0 = tid >> 2, r1 = (tid >> 2) + 64, cc = (tid & 3) * 8;
  auto stage = [&](int buf, int k0) {
    cp16(&As[buf][r0 * 32 + cc], Ag + (size_t)(m0 + r0) * K + k0 + cc);
    cp16(&As[buf][r1 * 32 + cc], Ag + (size_t)(m0 + r1) * K + k0 + cc);
    cp16(&Bs[buf][r0 * 32 + cc], Bt + (size_t)(n0 + r0) * K + k0 + cc);
    cp16(&Bs[buf][r1 * 32 + cc], Bt + (size_t)(n0 + r1) * K + k0 + cc);
  };
  v8f acc[2][4];
#pragma unroll
  for (int i = 0; i < 2; ++i)
#pragma unroll
    for (int j = 0; j < 4; ++j) acc[i][j] = vzero();
  stage(0, 0);
  wait_cp();
  __syncthreads();
  const int nk = K >> 5;
  for (int kt = 0; kt < nk; ++kt) {
    const int cur = kt & 1;
    if (kt + 1 < nk) stage(cur ^ 1, (kt + 1) * 32);
    if (kt + 2 < nk) __builtin_prefetch(Ag + (size_t)(m0 + (tid >> 1)) * K + (kt + 2) * 32, 0, 1);
    v16bf av[2], bv[4];
#pragma unroll
    for (int i = 0; i < 2; ++i) {
      const unsigned short* p = &As[cur][(wm * 32 + i * 16 + ln) * 32];
      av[i] = ldfrag(p + 16 * half, p + 16 * half + 8);
    }
#pragma unroll
    for (int j = 0; j < 4; ++j) {
      const unsigned short* p = &Bs[cur][(wn * 64 + j * 16 + ln) * 32 + 16 * half];
      bv[j] = ldfrag(p, p + 8);
    }
#pragma unroll
    for (int i = 0; i < 2; ++i)
#pragma unroll
      for (int j = 0; j < 4; ++j) acc[i][j] = wmma_bf16(av[i], bv[j], acc[i][j]);
    wait_cp();
    __syncthreads();
  }
#pragma unroll
  for (int i = 0; i < 2; ++i)
#pragma unroll
    for (int j = 0; j < 4; ++j)
#pragma unroll
      for (int r = 0; r < 8; ++r) {
        int row = m0 + wm * 32 + i * 16 + r + 8 * half;
        int col = n0 + wn * 64 + j * 16 + ln;
        C[(size_t)row * N + col] = acc[i][j][r];
      }
}

// ---------------- gather + GLU fused GEMM (shared expert & MoE up-proj) ----
// ACT[row][n] = silu(A@W1[n]) * (A@W3[n]); Bt rows [0,I) = gate, [I,2I) = up.
__global__ __launch_bounds__(256) void k_glu(const unsigned short* __restrict__ Aall,
                                             const unsigned short* __restrict__ Ball,
                                             unsigned short* __restrict__ ACTall,
                                             const int* __restrict__ bidx_all,
                                             const int* __restrict__ bcnt,
                                             int K, int I, unsigned long long strideB,
                                             unsigned long long strideACT) {
  const int e = blockIdx.z;
  const unsigned short* Bg = Ball + (size_t)e * strideB;
  unsigned short* ACT = ACTall + (size_t)e * strideACT;
  const int* gidx = bidx_all ? (bidx_all + e * T_TOK) : nullptr;
  const int cnt = bcnt ? bcnt[e] : (int)(gridDim.y * 128);
  const int m0 = blockIdx.y * 128, n0 = blockIdx.x * 64;
  if (m0 >= cnt) return;
  __shared__ unsigned short As[2][128 * 32];
  __shared__ unsigned short Bsg[2][64 * 32];
  __shared__ unsigned short Bsu[2][64 * 32];
  const int tid = threadIdx.x, lane = tid & 31, wave = tid >> 5;
  const int ln = lane & 15, half = lane >> 4;
  const int wm = wave >> 1, wn = wave & 1;
  const int ra0 = tid >> 2, ra1 = (tid >> 2) + 64, cc = (tid & 3) * 8;
  const int gm0 = m0 + ra0, gm1 = m0 + ra1;
  const int tok0 = gidx ? ((gm0 < cnt) ? gidx[gm0] : 0) : gm0;
  const int tok1 = gidx ? ((gm1 < cnt) ? gidx[gm1] : 0) : gm1;
  const int rb = tid >> 2;
  auto stage = [&](int buf, int k0) {
    cp16(&As[buf][ra0 * 32 + cc], Aall + (size_t)tok0 * K + k0 + cc);
    cp16(&As[buf][ra1 * 32 + cc], Aall + (size_t)tok1 * K + k0 + cc);
    cp16(&Bsg[buf][rb * 32 + cc], Bg + (size_t)(n0 + rb) * K + k0 + cc);
    cp16(&Bsu[buf][rb * 32 + cc], Bg + (size_t)(I + n0 + rb) * K + k0 + cc);
  };
  v8f ag[2][2], au[2][2];
#pragma unroll
  for (int i = 0; i < 2; ++i)
#pragma unroll
    for (int j = 0; j < 2; ++j) { ag[i][j] = vzero(); au[i][j] = vzero(); }
  stage(0, 0);
  wait_cp();
  __syncthreads();
  const int nk = K >> 5;
  for (int kt = 0; kt < nk; ++kt) {
    const int cur = kt & 1;
    if (kt + 1 < nk) stage(cur ^ 1, (kt + 1) * 32);
    v16bf av[2], bg[2], bu[2];
#pragma unroll
    for (int i = 0; i < 2; ++i) {
      const unsigned short* p = &As[cur][(wm * 32 + i * 16 + ln) * 32];
      av[i] = ldfrag(p + 16 * half, p + 16 * half + 8);
    }
#pragma unroll
    for (int j = 0; j < 2; ++j) {
      const unsigned short* pg = &Bsg[cur][(wn * 32 + j * 16 + ln) * 32 + 16 * half];
      const unsigned short* pu = &Bsu[cur][(wn * 32 + j * 16 + ln) * 32 + 16 * half];
      bg[j] = ldfrag(pg, pg + 8);
      bu[j] = ldfrag(pu, pu + 8);
    }
#pragma unroll
    for (int i = 0; i < 2; ++i)
#pragma unroll
      for (int j = 0; j < 2; ++j) {
        ag[i][j] = wmma_bf16(av[i], bg[j], ag[i][j]);
        au[i][j] = wmma_bf16(av[i], bu[j], au[i][j]);
      }
    wait_cp();
    __syncthreads();
  }
#pragma unroll
  for (int i = 0; i < 2; ++i)
#pragma unroll
    for (int j = 0; j < 2; ++j)
#pragma unroll
      for (int r = 0; r < 8; ++r) {
        int gm = m0 + wm * 32 + i * 16 + r + 8 * half;
        if (gm < cnt) {
          int col = n0 + wn * 32 + j * 16 + ln;
          float g = ag[i][j][r], u = au[i][j][r];
          float act = g / (1.f + __expf(-g)) * u;
          ACT[(size_t)gm * I + col] = f2bf(act);
        }
      }
}

// ---------------- MoE down-proj GEMM with weighted scatter into partials ---
__global__ __launch_bounds__(256) void k_scatter(const unsigned short* __restrict__ Aall,
                                                 const unsigned short* __restrict__ Ball,
                                                 float* __restrict__ part,
                                                 const int* __restrict__ bidx,
                                                 const int* __restrict__ bslot,
                                                 const float* __restrict__ bwgt,
                                                 const int* __restrict__ bcnt,
                                                 int K, int N,
                                                 unsigned long long strideA,
                                                 unsigned long long strideB) {
  const int e = blockIdx.z;
  const int cnt = bcnt[e];
  const int m0 = blockIdx.y * 128, n0 = blockIdx.x * 128;
  if (m0 >= cnt) return;
  const unsigned short* Ag = Aall + (size_t)e * strideA;
  const unsigned short* Bt = Ball + (size_t)e * strideB;
  __shared__ unsigned short As[2][128 * 32];
  __shared__ unsigned short Bs[2][128 * 32];
  const int tid = threadIdx.x, lane = tid & 31, wave = tid >> 5;
  const int ln = lane & 15, half = lane >> 4;
  const int wm = wave >> 1, wn = wave & 1;
  const int r0 = tid >> 2, r1 = (tid >> 2) + 64, cc = (tid & 3) * 8;
  auto stage = [&](int buf, int k0) {
    cp16(&As[buf][r0 * 32 + cc], Ag + (size_t)(m0 + r0) * K + k0 + cc);
    cp16(&As[buf][r1 * 32 + cc], Ag + (size_t)(m0 + r1) * K + k0 + cc);
    cp16(&Bs[buf][r0 * 32 + cc], Bt + (size_t)(n0 + r0) * K + k0 + cc);
    cp16(&Bs[buf][r1 * 32 + cc], Bt + (size_t)(n0 + r1) * K + k0 + cc);
  };
  v8f acc[2][4];
#pragma unroll
  for (int i = 0; i < 2; ++i)
#pragma unroll
    for (int j = 0; j < 4; ++j) acc[i][j] = vzero();
  stage(0, 0);
  wait_cp();
  __syncthreads();
  const int nk = K >> 5;
  for (int kt = 0; kt < nk; ++kt) {
    const int cur = kt & 1;
    if (kt + 1 < nk) stage(cur ^ 1, (kt + 1) * 32);
    v16bf av[2], bv[4];
#pragma unroll
    for (int i = 0; i < 2; ++i) {
      const unsigned short* p = &As[cur][(wm * 32 + i * 16 + ln) * 32];
      av[i] = ldfrag(p + 16 * half, p + 16 * half + 8);
    }
#pragma unroll
    for (int j = 0; j < 4; ++j) {
      const unsigned short* p = &Bs[cur][(wn * 64 + j * 16 + ln) * 32 + 16 * half];
      bv[j] = ldfrag(p, p + 8);
    }
#pragma unroll
    for (int i = 0; i < 2; ++i)
#pragma unroll
      for (int j = 0; j < 4; ++j) acc[i][j] = wmma_bf16(av[i], bv[j], acc[i][j]);
    wait_cp();
    __syncthreads();
  }
#pragma unroll
  for (int i = 0; i < 2; ++i)
#pragma unroll
    for (int j = 0; j < 4; ++j)
#pragma unroll
      for (int r = 0; r < 8; ++r) {
        int gm = m0 + wm * 32 + i * 16 + r + 8 * half;
        if (gm < cnt) {
          int tok = bidx[e * T_TOK + gm];
          int sl  = bslot[e * T_TOK + gm];
          float w = bwgt[e * T_TOK + gm];
          int col = n0 + wn * 64 + j * 16 + ln;
          part[((size_t)tok * 4 + sl) * HDIM + col] = w * acc[i][j][r];
        }
      }
}

// ---------------- RoPE + pack q/k/v to bf16 attention layouts --------------
// qb: [NH][T][HD], kb: [NKV][T][HD], vtb: [NKV][HD][T] (transposed)
__global__ __launch_bounds__(256) void k_rope(const float* __restrict__ qkv,
                                              const int* __restrict__ pos_ids,
                                              unsigned short* __restrict__ qb,
                                              unsigned short* __restrict__ kb,
                                              unsigned short* __restrict__ vtb) {
  const int t = blockIdx.x, tid = threadIdx.x;
  const float pos = (float)pos_ids[t];
#pragma unroll
  for (int i = 0; i < 4; ++i) {   // q: 16 heads x 64 pairs
    int p = tid + i * 256;
    int hd = p >> 6, d2 = p & 63;
    float inv = __expf(-(float)(2 * d2) * (9.210340371976184f / 128.f));
    float ang = pos * inv, c = __cosf(ang), s = __sinf(ang);
    float x1 = qkv[(size_t)t * QKVN + hd * HD + d2];
    float x2 = qkv[(size_t)t * QKVN + hd * HD + d2 + 64];
    size_t base = ((size_t)hd * T_TOK + t) * HD;
    qb[base + d2]      = f2bf(x1 * c - x2 * s);
    qb[base + d2 + 64] = f2bf(x2 * c + x1 * s);
  }
  { // k: 4 heads x 64 pairs
    int p = tid;
    int hd = p >> 6, d2 = p & 63;
    float inv = __expf(-(float)(2 * d2) * (9.210340371976184f / 128.f));
    float ang = pos * inv, c = __cosf(ang), s = __sinf(ang);
    float x1 = qkv[(size_t)t * QKVN + 2048 + hd * HD + d2];
    float x2 = qkv[(size_t)t * QKVN + 2048 + hd * HD + d2 + 64];
    size_t base = ((size_t)hd * T_TOK + t) * HD;
    kb[base + d2]      = f2bf(x1 * c - x2 * s);
    kb[base + d2 + 64] = f2bf(x2 * c + x1 * s);
  }
#pragma unroll
  for (int i = 0; i < 2; ++i) {   // v transposed
    int p = tid + i * 256;
    int hd = p >> 7, d = p & 127;
    vtb[((size_t)hd * HD + d) * T_TOK + t] =
        f2bf(qkv[(size_t)t * QKVN + 2560 + hd * HD + d]);
  }
}

// ---------------- WMMA flash attention: one wave per (head, 16-query tile) -
__global__ __launch_bounds__(32) void k_attn(const unsigned short* __restrict__ qb,
                                             const unsigned short* __restrict__ kb,
                                             const unsigned short* __restrict__ vtb,
                                             unsigned short* __restrict__ ctx) {
  const int qt = blockIdx.x, h = blockIdx.y;
  const int lane = threadIdx.x, ln = lane & 15, half = lane >> 4;
  const int qbase = qt * 16;
  const int kvh = h >> 2;
  __shared__ unsigned short Pld[16 * 32];
  v16bf qf[4];
  const unsigned short* qrow = qb + ((size_t)h * T_TOK + qbase + ln) * HD;
#pragma unroll
  for (int c = 0; c < 4; ++c)
    qf[c] = ldfrag(qrow + 32 * c + 16 * half, qrow + 32 * c + 16 * half + 8);
  v8f o[8];
#pragma unroll
  for (int i = 0; i < 8; ++i) o[i] = vzero();
  float mrow[8], lrow[8];
#pragma unroll
  for (int r = 0; r < 8; ++r) { mrow[r] = -1e30f; lrow[r] = 0.f; }
  const float scale = 0.088388347648318447f;   // 1/sqrt(128)
  for (int jb = 0; jb < qbase + 16; jb += 32) {
    v8f s0 = vzero(), s1 = vzero();
    const unsigned short* kr0 = kb + ((size_t)kvh * T_TOK + jb + ln) * HD;
    const unsigned short* kr1 = kr0 + 16 * HD;
#pragma unroll
    for (int c = 0; c < 4; ++c) {
      v16bf k0f = ldfrag(kr0 + 32 * c + 16 * half, kr0 + 32 * c + 16 * half + 8);
      v16bf k1f = ldfrag(kr1 + 32 * c + 16 * half, kr1 + 32 * c + 16 * half + 8);
      s0 = wmma_bf16(qf[c], k0f, s0);
      s1 = wmma_bf16(qf[c], k1f, s1);
    }
    float p0v[8], p1v[8];
#pragma unroll
    for (int r = 0; r < 8; ++r) {
      int qi = qbase + r + 8 * half;
      float a = s0[r] * scale; if (jb + ln > qi) a = -1e30f;
      float b = s1[r] * scale; if (jb + 16 + ln > qi) b = -1e30f;
      float loc = fmaxf(a, b);
#pragma unroll
      for (int ofs = 8; ofs >= 1; ofs >>= 1) loc = fmaxf(loc, __shfl_xor(loc, ofs));
      float mn = fmaxf(mrow[r], loc);
      float sc = __expf(mrow[r] - mn);
      float p0 = __expf(a - mn), p1 = __expf(b - mn);
      float rs = p0 + p1;
#pragma unroll
      for (int ofs = 8; ofs >= 1; ofs >>= 1) rs += __shfl_xor(rs, ofs);
      lrow[r] = lrow[r] * sc + rs;
      mrow[r] = mn;
#pragma unroll
      for (int nt = 0; nt < 8; ++nt) o[nt][r] *= sc;
      p0v[r] = p0; p1v[r] = p1;
    }
#pragma unroll
    for (int r = 0; r < 8; ++r) {
      Pld[(r + 8 * half) * 32 + ln]      = f2bf(p0v[r]);
      Pld[(r + 8 * half) * 32 + 16 + ln] = f2bf(p1v[r]);
    }
    __syncthreads();
    v16bf pf = ldfrag(&Pld[ln * 32 + 16 * half], &Pld[ln * 32 + 16 * half + 8]);
#pragma unroll
    for (int nt = 0; nt < 8; ++nt) {
      const unsigned short* vp =
          vtb + ((size_t)kvh * HD + nt * 16 + ln) * T_TOK + jb + 16 * half;
      v16bf vf = ldfrag(vp, vp + 8);
      o[nt] = wmma_bf16(pf, vf, o[nt]);
    }
    __syncthreads();
  }
#pragma unroll
  for (int nt = 0; nt < 8; ++nt)
#pragma unroll
    for (int r = 0; r < 8; ++r) {
      int row = qbase + r + 8 * half;
      ctx[(size_t)row * HDIM + h * HD + nt * 16 + ln] = f2bf(o[nt][r] / lrow[r]);
    }
}

// ---------------- router: softmax + top-4 + bucket build (1 wave / token) --
__global__ __launch_bounds__(32) void k_router(const unsigned short* __restrict__ x2b,
                                               const float* __restrict__ gw,
                                               int* counts, int* bidx, int* bslot,
                                               float* bwgt) {
  const int t = blockIdx.x, lane = threadIdx.x;
  float logit = -1e30f;
  if (lane < NEXP) {
    float s = 0.f;
    for (int k = 0; k < HDIM; ++k)
      s += bf2f(x2b[(size_t)t * HDIM + k]) * gw[(size_t)k * NEXP + lane];
    logit = s;
  }
  float mx = logit;
#pragma unroll
  for (int o = 8; o >= 1; o >>= 1) mx = fmaxf(mx, __shfl_xor(mx, o));
  float p = (lane < NEXP) ? __expf(logit - mx) : 0.f;
  float sm = p;
#pragma unroll
  for (int o = 8; o >= 1; o >>= 1) sm += __shfl_xor(sm, o);
  p /= sm;
  int alive = (lane < NEXP) ? 1 : 0;
  float myw = 0.f; int myslot = -1; float wsum = 0.f;
  for (int k = 0; k < 4; ++k) {
    float v = alive ? p : -1.f;
    int id = lane;
#pragma unroll
    for (int o = 8; o >= 1; o >>= 1) {
      float ov = __shfl_xor(v, o);
      int oid = __shfl_xor(id, o);
      if (ov > v || (ov == v && oid < id)) { v = ov; id = oid; }
    }
    wsum += v;
    if (lane == id && alive) { alive = 0; myslot = k; myw = p; }
  }
  if (myslot >= 0 && lane < NEXP) {
    int pos = atomicAdd(&counts[lane], 1);
    bidx[lane * T_TOK + pos] = t;
    bslot[lane * T_TOK + pos] = myslot;
    bwgt[lane * T_TOK + pos] = myw / wsum;
  }
}

// ---------------- deterministic final combine: out += sum of 4 MoE slots ---
__global__ __launch_bounds__(256) void k_combine(float* __restrict__ out,
                                                 const float* __restrict__ part) {
  size_t i = (size_t)blockIdx.x * 256 + threadIdx.x;
  int t = (int)(i >> 11), c = (int)(i & 2047);
  float s = out[i];
#pragma unroll
  for (int sl = 0; sl < 4; ++sl) s += part[((size_t)t * 4 + sl) * HDIM + c];
  out[i] = s;
}

// ---------------- host-side launch sequence ----------------
extern "C" void kernel_launch(void* const* d_in, const int* in_sizes, int n_in,
                              void* d_out, int out_size, void* d_ws, size_t ws_size,
                              hipStream_t stream) {
  const float* hs     = (const float*)d_in[0];
  const float* res    = (const float*)d_in[1];
  const float* rms1w  = (const float*)d_in[2];
  const float* rms2w  = (const float*)d_in[3];
  const float* wqkv   = (const float*)d_in[4];
  const float* wdense = (const float*)d_in[5];
  const float* gatew  = (const float*)d_in[6];
  const float* w13    = (const float*)d_in[7];
  const float* w2     = (const float*)d_in[8];
  const float* sw13   = (const float*)d_in[9];
  const float* sw2    = (const float*)d_in[10];
  const int*   posids = (const int*)d_in[11];
  float* out = (float*)d_out;

  char* base = (char*)d_ws;
  size_t off = 0;
  auto wsa = [&](size_t bytes) -> void* {
    void* r = base + off;
    off = (off + bytes + 255) & ~(size_t)255;
    return r;
  };
  float*          residf  = (float*)wsa((size_t)T_TOK * HDIM * 4);
  unsigned short* xb      = (unsigned short*)wsa((size_t)T_TOK * HDIM * 2);
  float*          qkvf    = (float*)wsa((size_t)T_TOK * QKVN * 4);
  unsigned short* qb      = (unsigned short*)wsa((size_t)NH * T_TOK * HD * 2);
  unsigned short* kb      = (unsigned short*)wsa((size_t)NKV * T_TOK * HD * 2);
  (void)wsa(65536);  // OOB-read slack for masked attention tail
  unsigned short* vtb     = (unsigned short*)wsa((size_t)NKV * HD * T_TOK * 2);
  (void)wsa(65536);
  unsigned short* ctxb    = (unsigned short*)wsa((size_t)T_TOK * HDIM * 2);
  float*          attnf   = (float*)wsa((size_t)T_TOK * HDIM * 4);
  unsigned short* x2b     = (unsigned short*)wsa((size_t)T_TOK * HDIM * 2);
  unsigned short* actsh   = (unsigned short*)wsa((size_t)T_TOK * SI * 2);
  unsigned short* wqkvt   = (unsigned short*)wsa((size_t)HDIM * QKVN * 2);
  unsigned short* wdenset = (unsigned short*)wsa((size_t)HDIM * HDIM * 2);
  unsigned short* sw13t   = (unsigned short*)wsa((size_t)HDIM * 2 * SI * 2);
  unsigned short* sw2t    = (unsigned short*)wsa((size_t)SI * HDIM * 2);
  unsigned short* w13t    = (unsigned short*)wsa((size_t)NEXP * HDIM * 2 * EI * 2);
  unsigned short* w2t     = (unsigned short*)wsa((size_t)NEXP * EI * HDIM * 2);
  unsigned short* actmoe  = (unsigned short*)wsa((size_t)NEXP * T_TOK * EI * 2);
  float*          part    = (float*)wsa((size_t)T_TOK * 4 * HDIM * 4);
  int*            counts  = (int*)wsa(NEXP * 4);
  int*            bidx    = (int*)wsa((size_t)NEXP * T_TOK * 4);
  int*            bslot   = (int*)wsa((size_t)NEXP * T_TOK * 4);
  float*          bwgt    = (float*)wsa((size_t)NEXP * T_TOK * 4);

  k_zero_i32<<<1, 32, 0, stream>>>(counts, NEXP);

  // pre-norm + residual, x -> bf16
  k_add_rms<<<T_TOK, 256, 0, stream>>>(hs, res, rms1w, residf, xb);

  // weights -> bf16, stored K-major (transposed) for contiguous async staging
  k_cvt_t<<<dim3(QKVN / 32, HDIM / 32, 1), 256, 0, stream>>>(wqkv, wqkvt, HDIM, QKVN);
  k_cvt_t<<<dim3(HDIM / 32, HDIM / 32, 1), 256, 0, stream>>>(wdense, wdenset, HDIM, HDIM);
  k_cvt_t<<<dim3(2 * SI / 32, HDIM / 32, 1), 256, 0, stream>>>(sw13, sw13t, HDIM, 2 * SI);
  k_cvt_t<<<dim3(HDIM / 32, SI / 32, 1), 256, 0, stream>>>(sw2, sw2t, SI, HDIM);
  k_cvt_t<<<dim3(2 * EI / 32, HDIM / 32, NEXP), 256, 0, stream>>>(w13, w13t, HDIM, 2 * EI);
  k_cvt_t<<<dim3(HDIM / 32, EI / 32, NEXP), 256, 0, stream>>>(w2, w2t, EI, HDIM);

  // QKV projection
  k_gemm<<<dim3(QKVN / 128, T_TOK / 128), 256, 0, stream>>>(xb, wqkvt, qkvf, QKVN, HDIM);
  // RoPE + attention-friendly bf16 layouts
  k_rope<<<T_TOK, 256, 0, stream>>>(qkvf, posids, qb, kb, vtb);
  // flash attention (WMMA)
  k_attn<<<dim3(T_TOK / 16, NH), 32, 0, stream>>>(qb, kb, vtb, ctxb);
  // output projection
  k_gemm<<<dim3(HDIM / 128, T_TOK / 128), 256, 0, stream>>>(ctxb, wdenset, attnf, HDIM, HDIM);

  // resid2 (second output) + x2 bf16
  k_add_rms<<<T_TOK, 256, 0, stream>>>(attnf, residf, rms2w,
                                       out + (size_t)T_TOK * HDIM, x2b);

  // shared expert: fused GLU then down-proj straight into the output buffer
  k_glu<<<dim3(SI / 64, T_TOK / 128, 1), 256, 0, stream>>>(
      x2b, sw13t, actsh, nullptr, nullptr, HDIM, SI, 0ull, 0ull);
  k_gemm<<<dim3(HDIM / 128, T_TOK / 128), 256, 0, stream>>>(actsh, sw2t, out, HDIM, SI);

  // router + top-4 bucket construction
  k_router<<<T_TOK, 32, 0, stream>>>(x2b, gatew, counts, bidx, bslot, bwgt);

  // MoE: gathered GLU up-proj (only routed tokens), then weighted scatter down-proj
  k_glu<<<dim3(EI / 64, T_TOK / 128, NEXP), 256, 0, stream>>>(
      x2b, w13t, actmoe, bidx, counts, HDIM, EI,
      (unsigned long long)HDIM * 2 * EI, (unsigned long long)T_TOK * EI);
  k_scatter<<<dim3(HDIM / 128, T_TOK / 128, NEXP), 256, 0, stream>>>(
      actmoe, w2t, part, bidx, bslot, bwgt, counts, EI, HDIM,
      (unsigned long long)T_TOK * EI, (unsigned long long)EI * HDIM);

  // out = shared + sum over 4 slots (deterministic, no float atomics)
  k_combine<<<(T_TOK * HDIM) / 256, 256, 0, stream>>>(out, part);
}
